// similarity_measure1_618475290900
// MI455X (gfx1250) — matrix operand order repaired and checked by reading, not compiled, not run on hardware
//
#include <hip/hip_runtime.h>

typedef __bf16 v16bf __attribute__((ext_vector_type(16)));
typedef __bf16 v2bf  __attribute__((ext_vector_type(2)));
typedef float  v8f   __attribute__((ext_vector_type(8)));

#define NEG_SLOPE 0.01f
#define PIX_PER_IMG 65536   // H*W = 256*256 (reference shapes)
#define C0 128              // input channels
#define HROWH 72            // bounce-buffer row stride in halfs (36 dwords, 144 B):
                            //  - rows stay 16B aligned for b128 fragment loads
                            //  - 36*px mod 64 banks are all distinct for px=0..15

__device__ __forceinline__ float leaky(float x) {
    // x>=0 -> x>=0.01x ; x<0 -> 0.01x>x  => leaky == max(x, 0.01x)
    return fmaxf(x, NEG_SLOPE * x);
}

// D = A(16x32 bf16) * B(32x16 bf16) + C, f32 accumulate
__device__ __forceinline__ v8f wmma_bf16(v16bf a, v16bf b, v8f c) {
    return __builtin_amdgcn_wmma_f32_16x16x32_bf16(false, a, false, b, (short)0, c,
                                                   false, false);
}

// Stage one layer's weights (O x C, row major f32) into LDS as WMMA A-fragments.
// Fragment = 512 bf16 (32 lanes x 16 elems), elems in contiguous K order (matches B):
//   K = 32*kt + 16*(lane>=16) + e
//   M = 16*ot + (lane%16);  out-of-range (o>=O || k>=C) zero-padded.
__device__ void stage_layer(__bf16* dst, const float* w, int O, int C,
                            int nKT, int nFrag, int tid, int nthreads) {
    int total = nFrag * 512;
    for (int i = tid; i < total; i += nthreads) {
        int frag = i >> 9;
        int rem  = i & 511;
        int lane = rem >> 4, e = rem & 15;
        int ot = frag / nKT, kt = frag % nKT;
        int j = e >> 1, h = e & 1, hi = lane >> 4;
        int k = 32 * kt + 16 * hi + e;
        int o = 16 * ot + (lane & 15);
        float v = (o < O && k < C) ? w[o * C + k] : 0.0f;
        dst[i] = (__bf16)v;
    }
}

__device__ __forceinline__ v16bf load_afrag(const __bf16* wlds, int fragIdx, int lane) {
    return *reinterpret_cast<const v16bf*>(wlds + fragIdx * 512 + lane * 16);
}

// B fragment (32 ch x 16 px) from the pixel-major bf16 bounce buffer:
// linear B layout (lane<16 -> K=e, lane>=16 -> K=16+e) == 16 contiguous halfs.
__device__ __forceinline__ v16bf load_bfrag(const __bf16* h, int kt, int n2, int lane) {
    int hi = lane >> 4, px = (lane & 15) + 16 * n2;
    return *reinterpret_cast<const v16bf*>(h + px * HROWH + 32 * kt + 16 * hi);
}

// Layer-1 B fragment straight from global x (channel stride = PIX_PER_IMG floats).
__device__ __forceinline__ v16bf load_bfrag_x(const float* xb, int kt, int n2, int lane) {
    int hi = lane >> 4, px = (lane & 15) + 16 * n2;
    const float* p = xb + (size_t)(32 * kt + 16 * hi) * PIX_PER_IMG + px;
    v16bf b;
#pragma unroll
    for (int e = 0; e < 16; ++e) b[e] = (__bf16)p[(size_t)e * PIX_PER_IMG];
    return b;
}

// D tile (16 out-ch x 16 px) -> bounce buffer, fused LeakyReLU, packed bf16 pairs.
// D layout: lane l, vgpr r holds M = r + 8*(l/16), N = l%16.
__device__ __forceinline__ void store_d(__bf16* h, int ot, int n2, int lane, v8f acc) {
    int hi = lane >> 4, px = (lane & 15) + 16 * n2;
    __bf16* row = h + px * HROWH + 16 * ot + 8 * hi;
#pragma unroll
    for (int i = 0; i < 4; ++i) {
        v2bf p = { (__bf16)leaky(acc[2 * i]), (__bf16)leaky(acc[2 * i + 1]) };
        *reinterpret_cast<v2bf*>(row + 2 * i) = p;   // 4B packed store
    }
}

// Layers 5-7 (8->4->2->1) on lanes 0-15: lane holds all 8 channels of its pixel.
__device__ __forceinline__ float tail_chain(const v8f& acc4, const float* W5,
                                            const float* W6, const float* W7) {
    float h4[8];
#pragma unroll
    for (int r = 0; r < 8; ++r) h4[r] = leaky(acc4[r]);
    float h5[4];
#pragma unroll
    for (int o = 0; o < 4; ++o) {
        float s = 0.0f;
#pragma unroll
        for (int c = 0; c < 8; ++c) s += W5[o * 8 + c] * h4[c];
        h5[o] = leaky(s);
    }
    float h6[2];
#pragma unroll
    for (int o = 0; o < 2; ++o) {
        float s = 0.0f;
#pragma unroll
        for (int c = 0; c < 4; ++c) s += W6[o * 4 + c] * h5[c];
        h6[o] = leaky(s);
    }
    return W7[0] * h6[0] + W7[1] * h6[1];
}

__global__ __launch_bounds__(256) void conv1x1_chain_kernel(
    const float* __restrict__ x,
    const float* __restrict__ w1, const float* __restrict__ w2,
    const float* __restrict__ w3, const float* __restrict__ w4,
    const float* __restrict__ w5, const float* __restrict__ w6,
    const float* __restrict__ w7,
    float* __restrict__ out, int nTiles)
{
    // Weight fragments: L1 16 | L2 4 | L3 1 | L4 1 -> 22 * 512 bf16 = 22 KB
    __shared__ __align__(64) __bf16 wlds[22 * 512];
    // Per-wave pixel-major bounce: 32 px rows x 72 halfs = 4.5 KB/wave
    __shared__ __align__(64) __bf16 hbuf_all[8][32 * HROWH];

    const int tid  = threadIdx.x;
    const int lane = tid & 31;
    const int wave = tid >> 5;
    __bf16* hbuf = hbuf_all[wave];

    stage_layer(wlds + 0,        w1, 64, 128, 4, 16, tid, 256);
    stage_layer(wlds + 16 * 512, w2, 32,  64, 2,  4, tid, 256);
    stage_layer(wlds + 20 * 512, w3, 16,  32, 1,  1, tid, 256);
    stage_layer(wlds + 21 * 512, w4,  8,  16, 1,  1, tid, 256);
    __syncthreads();

    // Tail-layer weights: uniform, hoisted (scalar loads, live in SGPRs).
    float W5[32], W6[8], W7[2];
#pragma unroll
    for (int i = 0; i < 32; ++i) W5[i] = w5[i];
#pragma unroll
    for (int i = 0; i < 8; ++i)  W6[i] = w6[i];
    W7[0] = w7[0]; W7[1] = w7[1];

    const int waveStride = gridDim.x * 8;
    for (int tile = blockIdx.x * 8 + wave; tile < nTiles; tile += waveStride) {
        const int p0  = tile * 32;
        const int b   = p0 / PIX_PER_IMG;
        const int hw0 = p0 - b * PIX_PER_IMG;
        const float* xb = x + (size_t)b * C0 * PIX_PER_IMG + hw0;

        // ---- Layer 1 loads: 128 ch x 32 px, direct global -> bf16 fragments ----
        v16bf bx[4][2];
#pragma unroll
        for (int kt = 0; kt < 4; ++kt) {
            bx[kt][0] = load_bfrag_x(xb, kt, 0, lane);
            bx[kt][1] = load_bfrag_x(xb, kt, 1, lane);
        }

        // Prefetch next tile's 128 channel rows (one 128B line each) into cache.
        {
            int nt = tile + waveStride;
            if (nt < nTiles) {
                const int np0  = nt * 32;
                const int nb   = np0 / PIX_PER_IMG;
                const int nhw0 = np0 - nb * PIX_PER_IMG;
                const float* xn = x + (size_t)nb * C0 * PIX_PER_IMG + nhw0;
#pragma unroll
                for (int q = 0; q < 4; ++q)
                    __builtin_prefetch(xn + (size_t)(q * 32 + lane) * PIX_PER_IMG, 0, 1);
            }
        }

        // ---- Layer 1: 128 -> 64 ----
#pragma unroll
        for (int ot = 0; ot < 4; ++ot) {
            v8f acc0 = {}, acc1 = {};
#pragma unroll
            for (int kt = 0; kt < 4; ++kt) {
                v16bf a = load_afrag(wlds, ot * 4 + kt, lane);
                acc0 = wmma_bf16(a, bx[kt][0], acc0);
                acc1 = wmma_bf16(a, bx[kt][1], acc1);
            }
            store_d(hbuf, ot, 0, lane, acc0);
            store_d(hbuf, ot, 1, lane, acc1);
        }

        // ---- Layer 2: 64 -> 32 ----
        {
            v16bf b00 = load_bfrag(hbuf, 0, 0, lane);
            v16bf b01 = load_bfrag(hbuf, 0, 1, lane);
            v16bf b10 = load_bfrag(hbuf, 1, 0, lane);
            v16bf b11 = load_bfrag(hbuf, 1, 1, lane);
#pragma unroll
            for (int ot = 0; ot < 2; ++ot) {
                v16bf a0 = load_afrag(wlds, 16 + ot * 2 + 0, lane);
                v16bf a1 = load_afrag(wlds, 16 + ot * 2 + 1, lane);
                v8f acc0 = {}, acc1 = {};
                acc0 = wmma_bf16(a0, b00, acc0);
                acc0 = wmma_bf16(a1, b10, acc0);
                acc1 = wmma_bf16(a0, b01, acc1);
                acc1 = wmma_bf16(a1, b11, acc1);
                store_d(hbuf, ot, 0, lane, acc0);
                store_d(hbuf, ot, 1, lane, acc1);
            }
        }

        // ---- Layer 3: 32 -> 16 ----
        {
            v16bf b0 = load_bfrag(hbuf, 0, 0, lane);
            v16bf b1 = load_bfrag(hbuf, 0, 1, lane);
            v16bf a  = load_afrag(wlds, 20, lane);
            v8f acc0 = {}, acc1 = {};
            acc0 = wmma_bf16(a, b0, acc0);
            acc1 = wmma_bf16(a, b1, acc1);
            store_d(hbuf, 0, 0, lane, acc0);
            store_d(hbuf, 0, 1, lane, acc1);
        }

        // ---- Layer 4: 16 -> 8 (A has zeros for K>=16 and M>=8, so the stale
        //      upper-half B values are multiplied by zero; all data is finite) ----
        v8f acc4a, acc4b;
        {
            v16bf b0 = load_bfrag(hbuf, 0, 0, lane);
            v16bf b1 = load_bfrag(hbuf, 0, 1, lane);
            v16bf a  = load_afrag(wlds, 21, lane);
            v8f c0 = {}, c1 = {};
            acc4a = wmma_bf16(a, b0, c0);
            acc4b = wmma_bf16(a, b1, c1);
        }

        // ---- Layers 5-7: VALU on lanes 0-15 (each lane owns 2 pixels) ----
        if (lane < 16) {
            out[p0 + lane]      = tail_chain(acc4a, W5, W6, W7);
            out[p0 + 16 + lane] = tail_chain(acc4b, W5, W6, W7);
        }
    }
}

extern "C" void kernel_launch(void* const* d_in, const int* in_sizes, int n_in,
                              void* d_out, int out_size, void* d_ws, size_t ws_size,
                              hipStream_t stream) {
    const float* x  = (const float*)d_in[0];
    const float* w1 = (const float*)d_in[1];
    const float* w2 = (const float*)d_in[2];
    const float* w3 = (const float*)d_in[3];
    const float* w4 = (const float*)d_in[4];
    const float* w5 = (const float*)d_in[5];
    const float* w6 = (const float*)d_in[6];
    const float* w7 = (const float*)d_in[7];
    float* out = (float*)d_out;

    const int pixels = in_sizes[0] / C0;   // B*H*W
    const int nTiles = pixels / 32;        // 16384 for reference shapes

    conv1x1_chain_kernel<<<dim3(1024), dim3(256), 0, stream>>>(
        x, w1, w2, w3, w4, w5, w6, w7, out, nTiles);
}
